// _NonLocalBlockND_38568806318678
// MI455X (gfx1250) — matrix operand order repaired and checked by reading, not compiled, not run on hardware
//
#include <hip/hip_runtime.h>
#include <hip/hip_bf16.h>

// NonLocal block (self-attention) forward for MI455X (gfx1250, wave32, WMMA).
// B=4, C=CI=512, L=2048, H=8, D=64.
// Strategy: bf16 WMMA 16x16x32 with f32 accumulation for all four GEMMs,
// flash-style online softmax, BN via atomically-reduced channel statistics.

#define NB 4
#define NC 512
#define NL 2048
#define NCI 512
#define NH 8
#define ND 64
#define BN_EPS 1e-5f

typedef __bf16 bf16;
typedef __attribute__((ext_vector_type(16))) __bf16 bf16x16;
typedef __attribute__((ext_vector_type(8)))  float  f32x8;

union Frag { bf16x16 v; uint4 q[2]; };
union Acc  { f32x8  v; float f[8]; };

__device__ __forceinline__ f32x8 wmma_bf16(bf16x16 a, bf16x16 b, f32x8 c) {
  // D = A(16x32) * B(32x16) + C, f32 accumulate
  return __builtin_amdgcn_wmma_f32_16x16x32_bf16(false, a, false, b, (short)0, c,
                                                 false, false);
}

// A-fragment (16x32 bf16) from row-major [M][K] storage, K contiguous.
// ISA 7.12.2: lanes 0-15 hold row r with K=0..7 (v0..3) and K=16..23 (v4..7);
// lanes 16-31 hold row r with K=8..15 and K=24..31.  => two 16B loads/lane.
__device__ __forceinline__ bf16x16 load_a_frag(const bf16* base, int ldk, int lane) {
  const int r = lane & 15, h = lane >> 4;
  Frag u;
  u.q[0] = *reinterpret_cast<const uint4*>(base + (size_t)r * ldk + h * 8);
  u.q[1] = *reinterpret_cast<const uint4*>(base + (size_t)r * ldk + 16 + h * 8);
  return u.v;
}

// B-fragment (32x16 bf16) from [N][K] storage, K contiguous.
// lanes 0-15: column n=r, K=0..7 and K=16..23; lanes 16-31: column n=r, K=8..15 and K=24..31.
__device__ __forceinline__ bf16x16 load_b_frag(const bf16* base, int ldk, int lane) {
  const int r = lane & 15, h = lane >> 4;
  const uint4* p = reinterpret_cast<const uint4*>(base + (size_t)r * ldk + h * 8);
  Frag u;
  u.q[0] = p[0];
  u.q[1] = p[2];
  return u.v;
}

// ---------------- prep kernels ----------------

__global__ void k_cvt_bf16(const float* __restrict__ in, bf16* __restrict__ out, int n) {
  int i = blockIdx.x * blockDim.x + threadIdx.x;
  if (i < n) out[i] = (bf16)in[i];
}

// x [B,C,L] f32 -> xT [B,L,C] bf16 (K-contiguous for projection B-fragments)
__global__ void k_xT(const float* __restrict__ x, bf16* __restrict__ xT) {
  int i = blockIdx.x * blockDim.x + threadIdx.x;
  if (i >= NB * NC * NL) return;
  int l = i % NL;
  int c = (i / NL) % NC;
  int b = i / (NL * NC);
  xT[((size_t)b * NL + l) * NC + c] = (bf16)x[i];
}

// ---------------- projection GEMM (theta / phi / g) ----------------
// out[o,n] = sum_c W[o,c] * x[b,c,n] + bias[o];  M=CI, N=L, K=C per batch.
// grid (NL/64, NCI/64, NB*3), block 128 (4 waves, 16 rows x 64 cols each).
__global__ void k_proj(const bf16* __restrict__ xT,
                       const bf16* __restrict__ wt, const bf16* __restrict__ wp,
                       const bf16* __restrict__ wg,
                       const float* __restrict__ bt, const float* __restrict__ bp,
                       const float* __restrict__ bg,
                       bf16* __restrict__ theta_t, bf16* __restrict__ phi_t,
                       bf16* __restrict__ g_n) {
  const int lane = threadIdx.x & 31;
  const int wave = threadIdx.x >> 5;
  const int z = blockIdx.z;
  const int b = z / 3, p = z % 3;
  const bf16*  W    = (p == 0) ? wt : (p == 1) ? wp : wg;
  const float* bias = (p == 0) ? bt : (p == 1) ? bp : bg;

  const int o0 = blockIdx.y * 64 + wave * 16;   // output channels (rows)
  const int n0 = blockIdx.x * 64;               // l positions (cols)
  const bf16* xTb = xT + (size_t)b * NL * NC;   // [L][C]

  Acc acc[4];
#pragma unroll
  for (int t = 0; t < 4; ++t)
#pragma unroll
    for (int j = 0; j < 8; ++j) acc[t].f[j] = 0.0f;

  for (int k = 0; k < NC; k += 32) {
    bf16x16 a = load_a_frag(W + (size_t)o0 * NC + k, NC, lane);
#pragma unroll
    for (int t = 0; t < 4; ++t) {
      bf16x16 bb = load_b_frag(xTb + (size_t)(n0 + t * 16) * NC + k, NC, lane);
      acc[t].v = wmma_bf16(a, bb, acc[t].v);
    }
  }

  const int r = lane & 15, h = lane >> 4;
#pragma unroll
  for (int t = 0; t < 4; ++t) {
    const int n = n0 + t * 16 + r;
#pragma unroll
    for (int j = 0; j < 8; ++j) {
      const int o = o0 + j + 8 * h;
      const bf16 bv = (bf16)(acc[t].f[j] + bias[o]);
      if (p == 2) {
        g_n[((size_t)b * NCI + o) * NL + n] = bv;                    // [B,H,D,L]
      } else {
        bf16* dst = (p == 0) ? theta_t : phi_t;                      // [B,H,L,D]
        dst[(((size_t)b * NH + (o >> 6)) * NL + n) * ND + (o & 63)] = bv;
      }
    }
  }
}

// ---------------- flash attention ----------------
// grid (NL/64, NH, NB), block 128; each wave owns 16 queries x D=64.
__global__ void k_attn(const bf16* __restrict__ theta_t, const bf16* __restrict__ phi_t,
                       const bf16* __restrict__ g_n, bf16* __restrict__ yT) {
  __shared__ alignas(16) bf16 Pl[4][16][64];   // per-wave P tile (row-major)

  const int lane = threadIdx.x & 31;
  const int wave = threadIdx.x >> 5;
  const int b = blockIdx.z, hh = blockIdx.y;
  const int q0 = blockIdx.x * 64 + wave * 16;

  const bf16* th = theta_t + ((size_t)b * NH + hh) * NL * ND;  // [L][D]
  const bf16* ph = phi_t   + ((size_t)b * NH + hh) * NL * ND;  // [L][D]
  const bf16* gg = g_n     + ((size_t)b * NH + hh) * ND * NL;  // [D][L]

  // theta A-fragments for this wave's 16 query rows (K = d: 0..31, 32..63)
  const bf16x16 a0 = load_a_frag(th + (size_t)q0 * ND, ND, lane);
  const bf16x16 a1 = load_a_frag(th + (size_t)q0 * ND + 32, ND, lane);

  Acc o_acc[4];
  float m[8], ssum[8];
#pragma unroll
  for (int t = 0; t < 4; ++t)
#pragma unroll
    for (int j = 0; j < 8; ++j) o_acc[t].f[j] = 0.0f;
#pragma unroll
  for (int j = 0; j < 8; ++j) { m[j] = -1e30f; ssum[j] = 0.0f; }

  const float scale = 0.125f;  // 1/sqrt(64)
  const int r = lane & 15, h = lane >> 4;

  for (int kt = 0; kt < NL; kt += 64) {
    // ---- scores S = theta * phi^T over this 64-key tile ----
    Acc s[4];
#pragma unroll
    for (int t = 0; t < 4; ++t) {
#pragma unroll
      for (int j = 0; j < 8; ++j) s[t].f[j] = 0.0f;
      bf16x16 b0 = load_b_frag(ph + (size_t)(kt + t * 16) * ND, ND, lane);
      bf16x16 b1 = load_b_frag(ph + (size_t)(kt + t * 16) * ND + 32, ND, lane);
      s[t].v = wmma_bf16(a0, b0, s[t].v);
      s[t].v = wmma_bf16(a1, b1, s[t].v);
    }

    // ---- online softmax: row stats live per C-fragment register j ----
#pragma unroll
    for (int j = 0; j < 8; ++j) {
      float mx = fmaxf(fmaxf(s[0].f[j], s[1].f[j]), fmaxf(s[2].f[j], s[3].f[j])) * scale;
#pragma unroll
      for (int d = 1; d < 16; d <<= 1) mx = fmaxf(mx, __shfl_xor(mx, d, 32));
      const float mnew = fmaxf(m[j], mx);
      const float corr = __expf(m[j] - mnew);
      ssum[j] *= corr;
#pragma unroll
      for (int t = 0; t < 4; ++t) o_acc[t].f[j] *= corr;
      m[j] = mnew;
    }

    // ---- P = exp(S*scale - m): sum rows, stage in LDS row-major ----
    float psum[8];
#pragma unroll
    for (int j = 0; j < 8; ++j) psum[j] = 0.0f;
#pragma unroll
    for (int t = 0; t < 4; ++t) {
#pragma unroll
      for (int j = 0; j < 8; ++j) {
        const float pv = __expf(s[t].f[j] * scale - m[j]);
        psum[j] += pv;
        Pl[wave][j + 8 * h][t * 16 + r] = (bf16)pv;
      }
    }
#pragma unroll
    for (int j = 0; j < 8; ++j) {
      float ps = psum[j];
#pragma unroll
      for (int d = 1; d < 16; d <<= 1) ps += __shfl_xor(ps, d, 32);
      ssum[j] += ps;
    }
    __syncthreads();

    // ---- O += P * g : P A-fragments from LDS, g B-fragments from global ----
    const bf16* pl = &Pl[wave][0][0];
    bf16x16 pa0 = load_a_frag(pl, 64, lane);        // K = kt+0..31
    bf16x16 pa1 = load_a_frag(pl + 32, 64, lane);   // K = kt+32..63
#pragma unroll
    for (int t = 0; t < 4; ++t) {
      bf16x16 g0 = load_b_frag(gg + (size_t)(t * 16) * NL + kt, NL, lane);
      bf16x16 g1 = load_b_frag(gg + (size_t)(t * 16) * NL + kt + 32, NL, lane);
      o_acc[t].v = wmma_bf16(pa0, g0, o_acc[t].v);
      o_acc[t].v = wmma_bf16(pa1, g1, o_acc[t].v);
    }
    __syncthreads();
  }

  // ---- normalize and emit y^T [B,L,CI] bf16 (K-contiguous for final conv) ----
#pragma unroll
  for (int t = 0; t < 4; ++t) {
#pragma unroll
    for (int j = 0; j < 8; ++j) {
      const int q = q0 + j + 8 * h;
      const int d = t * 16 + r;
      yT[((size_t)b * NL + q) * NCI + hh * ND + d] = (bf16)(o_acc[t].f[j] / ssum[j]);
    }
  }
}

// ---------------- final conv + BN partial sums ----------------
// wy[co,n] = sum_ci Ww[co,ci] * y[b,n,ci] + wb[co]; grid (NL/64, NC/64, NB).
__global__ void k_wconv(const bf16* __restrict__ yT, const bf16* __restrict__ ww,
                        const float* __restrict__ wb, float* __restrict__ wy,
                        float* __restrict__ sum, float* __restrict__ sumsq) {
  const int lane = threadIdx.x & 31;
  const int wave = threadIdx.x >> 5;
  const int b = blockIdx.z;
  const int co0 = blockIdx.y * 64 + wave * 16;
  const int n0 = blockIdx.x * 64;
  const bf16* yTb = yT + (size_t)b * NL * NCI;   // [L][CI]

  Acc acc[4];
#pragma unroll
  for (int t = 0; t < 4; ++t)
#pragma unroll
    for (int j = 0; j < 8; ++j) acc[t].f[j] = 0.0f;

  for (int k = 0; k < NCI; k += 32) {
    bf16x16 a = load_a_frag(ww + (size_t)co0 * NCI + k, NCI, lane);
#pragma unroll
    for (int t = 0; t < 4; ++t) {
      bf16x16 bb = load_b_frag(yTb + (size_t)(n0 + t * 16) * NCI + k, NCI, lane);
      acc[t].v = wmma_bf16(a, bb, acc[t].v);
    }
  }

  const int r = lane & 15, h = lane >> 4;
  float rs[8], rs2[8];
#pragma unroll
  for (int j = 0; j < 8; ++j) { rs[j] = 0.0f; rs2[j] = 0.0f; }

#pragma unroll
  for (int t = 0; t < 4; ++t) {
    const int n = n0 + t * 16 + r;
#pragma unroll
    for (int j = 0; j < 8; ++j) {
      const int co = co0 + j + 8 * h;
      const float v = acc[t].f[j] + wb[co];
      wy[((size_t)b * NC + co) * NL + n] = v;
      rs[j] += v;
      rs2[j] += v * v;
    }
  }
#pragma unroll
  for (int j = 0; j < 8; ++j) {
    float a1 = rs[j], a2 = rs2[j];
#pragma unroll
    for (int d = 1; d < 16; d <<= 1) {
      a1 += __shfl_xor(a1, d, 32);
      a2 += __shfl_xor(a2, d, 32);
    }
    if (r == 0) {
      const int co = co0 + j + 8 * h;
      atomicAdd(&sum[co], a1);
      atomicAdd(&sumsq[co], a2);
    }
  }
}

// ---------------- BN finalize ----------------
__global__ void k_bn(const float* __restrict__ wy, const float* __restrict__ sum,
                     const float* __restrict__ sumsq, const float* __restrict__ gamma,
                     const float* __restrict__ beta, float* __restrict__ out) {
  int i = blockIdx.x * blockDim.x + threadIdx.x;
  if (i >= NB * NC * NL) return;
  const int c = (i / NL) % NC;
  const float inv_n = 1.0f / (float)(NB * NL);
  const float mean = sum[c] * inv_n;
  const float var = sumsq[c] * inv_n - mean * mean;
  out[i] = (wy[i] - mean) * rsqrtf(var + BN_EPS) * gamma[c] + beta[c];
}

extern "C" void kernel_launch(void* const* d_in, const int* in_sizes, int n_in,
                              void* d_out, int out_size, void* d_ws, size_t ws_size,
                              hipStream_t stream) {
  (void)in_sizes; (void)n_in; (void)out_size; (void)ws_size;
  const float* x    = (const float*)d_in[0];
  const float* g_w  = (const float*)d_in[1];
  const float* g_b  = (const float*)d_in[2];
  const float* th_w = (const float*)d_in[3];
  const float* th_b = (const float*)d_in[4];
  const float* ph_w = (const float*)d_in[5];
  const float* ph_b = (const float*)d_in[6];
  const float* w_w  = (const float*)d_in[7];
  const float* w_b  = (const float*)d_in[8];
  const float* gam  = (const float*)d_in[9];
  const float* bet  = (const float*)d_in[10];
  float* out = (float*)d_out;

  char* ws = (char*)d_ws;
  size_t off = 0;
  auto carve = [&](size_t bytes) -> char* {
    char* p = ws + off;
    off += (bytes + 255) & ~(size_t)255;
    return p;
  };
  bf16*  xT      = (bf16*)carve((size_t)NB * NL * NC * 2);
  bf16*  wt_bf   = (bf16*)carve((size_t)NCI * NC * 2);
  bf16*  wp_bf   = (bf16*)carve((size_t)NCI * NC * 2);
  bf16*  wg_bf   = (bf16*)carve((size_t)NCI * NC * 2);
  bf16*  ww_bf   = (bf16*)carve((size_t)NC * NCI * 2);
  bf16*  theta_t = (bf16*)carve((size_t)NB * NH * NL * ND * 2);
  bf16*  phi_t   = (bf16*)carve((size_t)NB * NH * NL * ND * 2);
  bf16*  g_nat   = (bf16*)carve((size_t)NB * NCI * NL * 2);
  bf16*  yT      = (bf16*)carve((size_t)NB * NL * NCI * 2);
  float* wy      = (float*)carve((size_t)NB * NC * NL * 4);
  float* sums    = (float*)carve((size_t)2 * NC * 4);  // sum | sumsq, adjacent

  const int nw = NCI * NC;
  k_cvt_bf16<<<(nw + 255) / 256, 256, 0, stream>>>(th_w, wt_bf, nw);
  k_cvt_bf16<<<(nw + 255) / 256, 256, 0, stream>>>(ph_w, wp_bf, nw);
  k_cvt_bf16<<<(nw + 255) / 256, 256, 0, stream>>>(g_w, wg_bf, nw);
  k_cvt_bf16<<<(nw + 255) / 256, 256, 0, stream>>>(w_w, ww_bf, nw);

  const int nx = NB * NC * NL;
  k_xT<<<(nx + 255) / 256, 256, 0, stream>>>(x, xT);

  k_proj<<<dim3(NL / 64, NCI / 64, NB * 3), 128, 0, stream>>>(
      xT, wt_bf, wp_bf, wg_bf, th_b, ph_b, g_b, theta_t, phi_t, g_nat);

  k_attn<<<dim3(NL / 64, NH, NB), 128, 0, stream>>>(theta_t, phi_t, g_nat, yT);

  hipMemsetAsync(sums, 0, (size_t)2 * NC * 4, stream);

  k_wconv<<<dim3(NL / 64, NC / 64, NB), 128, 0, stream>>>(
      yT, ww_bf, w_b, wy, sums, sums + NC);

  k_bn<<<(nx + 255) / 256, 256, 0, stream>>>(wy, sums, sums + NC, gam, bet, out);
}
